// AdaptiveGRU_71700184039675
// MI455X (gfx1250) — matrix operand order repaired and checked by reading, not compiled, not run on hardware
//
#include <hip/hip_runtime.h>
#include <hip/hip_bf16.h>

// ---------------------------------------------------------------------------
// AdaptiveGRU persistent kernel for gfx1250 (MI455X)
//   B=128, T=1024, F=128, H=256, O=128
//   All GEMMs via v_wmma_f32_16x16x32_f16 (16 WGs x 8 waves, wave32)
// ---------------------------------------------------------------------------

#define BB 128
#define TT 1024
#define FF 128
#define HH 256
#define OO 128
#define NWG 16
#define TEMP_INV 20.0f
#define EPS_SEL 1e-20f

typedef _Float16 half_t;
typedef __attribute__((ext_vector_type(16))) _Float16 v16h;
typedef __attribute__((ext_vector_type(8)))  _Float16 v8h;
typedef __attribute__((ext_vector_type(8)))  float    v8f;

union V16U { v16h v; v8h h[2]; };

// ---- WMMA wrapper ---------------------------------------------------------
__device__ __forceinline__ v8f wmma_f16(v16h a, v16h b, v8f c) {
  // D = A(16x32 f16) * B(32x16 f16) + C(16x16 f32)
  return __builtin_amdgcn_wmma_f32_16x16x32_f16(
      /*neg_a=*/false, a, /*neg_b=*/false, b,
      /*c_mod=*/(short)0, c, /*reuse_a=*/false, /*reuse_b=*/false);
}

// ---- A fragment: rows of a row-major [.., ld] f16 matrix ------------------
// Contiguous k order used here (matches B): lanes 0-15 hold K=0..15, lanes 16-31 K=16..31
__device__ __forceinline__ v16h load_a_frag(const half_t* base, int ld,
                                            int row0, int k0, int lane) {
  int m  = lane & 15;
  int ks = (lane >> 4) & 1;
  const half_t* p = base + (size_t)(row0 + m) * ld + k0 + ks * 16;
  V16U u;
  u.h[0] = *(const v8h*)(p);
  u.h[1] = *(const v8h*)(p + 8);
  return u.v;
}

// ---- B fragment: B = W^T, W row-major [.., ld]; lane's 16 K-halfs are
// contiguous in W's row (one aligned 32B load). Lanes 0-15: K=k0..k0+15,
// lanes 16-31: K=k0+16..k0+31, column = lane&15.
__device__ __forceinline__ v16h load_b_frag(const half_t* W, int ld,
                                            int col0, int k0, int lane) {
  const half_t* p = W + (size_t)(col0 + (lane & 15)) * ld + k0 + ((lane >> 4) & 1) * 16;
  return *(const v16h*)p;
}

__device__ __forceinline__ v8f splat8(float x) {
  v8f r;
#pragma unroll
  for (int i = 0; i < 8; ++i) r[i] = x;
  return r;
}

__device__ __forceinline__ float sigmoidf_(float x) {
  return 1.0f / (1.0f + __expf(-x));
}

// ---- grid-wide barrier (atomic count + generation) ------------------------
__device__ __forceinline__ void grid_barrier(int* cnt, int* gen) {
  __syncthreads();
  __threadfence();
  if (threadIdx.x == 0) {
    int g = atomicAdd(gen, 0);
    if (atomicAdd(cnt, 1) == NWG - 1) {
      atomicExch(cnt, 0);
      atomicAdd(gen, 1);
    } else {
      while (atomicAdd(gen, 0) == g) { __builtin_amdgcn_s_sleep(2); }
    }
  }
  __syncthreads();
  __threadfence();
}

// ---------------------------------------------------------------------------
// Init: f32->f16 weight conversion, zero h0, gate-0 input (w0 == 1), w0 out,
// barrier + num_selections reset. Re-runs identically every launch.
// ---------------------------------------------------------------------------
__global__ void adagru_init(const float* __restrict__ x,
                            const float* __restrict__ w_ih,
                            const float* __restrict__ w_hh,
                            const float* __restrict__ W_out,
                            const float* __restrict__ W_sel,
                            float* h32, half_t* hf16, half_t* xg16,
                            half_t* wih16, half_t* whh16,
                            half_t* wout16, half_t* wsel16,
                            float* d_out, int* bar) {
  size_t i = (size_t)blockIdx.x * blockDim.x + threadIdx.x;
  if (i < (size_t)BB * HH) {           // h ping buffer (parity 0) = 0
    h32[i]  = 0.0f;
    hf16[i] = (half_t)0.0f;
  }
  if (i < (size_t)BB * FF) {           // step-0 gated input: w0 = 1 -> x[:,0,:]
    size_t b = i / FF, f = i % FF;
    xg16[i] = (half_t)x[b * (size_t)TT * FF + f];
    // selection_weights[0,:,:] = 1
    d_out[(size_t)TT * BB * OO + 1 + i] = 1.0f;
  }
  if (i == 0) {
    d_out[(size_t)TT * BB * OO] = 0.0f;  // num_selections accumulator
    bar[0] = 0;
    bar[1] = 0;
  }
  if (i < (size_t)3 * HH * FF) wih16[i] = (half_t)w_ih[i];
  if (i < (size_t)3 * HH * HH) whh16[i] = (half_t)w_hh[i];
  if (i < (size_t)OO * HH) {
    wout16[i] = (half_t)W_out[i];
    wsel16[i] = (half_t)W_sel[i];
  }
}

// ---------------------------------------------------------------------------
// Persistent kernel: 16 WGs x 256 threads (8 waves), loops over all T steps.
// ---------------------------------------------------------------------------
__global__ void __launch_bounds__(256)
adagru_persistent(const float* __restrict__ x,
                  const float* __restrict__ noise,
                  const float* __restrict__ b_ih,
                  const float* __restrict__ b_hh,
                  const float* __restrict__ b_out,
                  const float* __restrict__ b_sel,
                  const half_t* __restrict__ wih16,
                  const half_t* __restrict__ whh16,
                  const half_t* __restrict__ wout16,
                  const half_t* __restrict__ wsel16,
                  float* h32, half_t* hf16, half_t* xg16,
                  float* d_out, int* bar) {
  const int wg   = blockIdx.x;          // 0..15  -> H column tile (phase A/B)
  const int wv   = threadIdx.x >> 5;    // 0..7   -> batch row tile
  const int lane = threadIdx.x & 31;
  const int rt   = wv;
  const int ct   = wg;
  const int ncol = lane & 15;
  const int hi8  = (lane >> 4) & 1;

  float* outputs = d_out;                                   // [T,B,O]
  float* nsel    = d_out + (size_t)TT * BB * OO;            // scalar
  float* selw    = nsel + 1;                                // [T,B,F]
  int* bcnt = bar;
  int* bgen = bar + 1;

  // --- per-lane gate biases (column = ct*16 + ncol of H) ---
  const int hc = ct * 16 + ncol;
  const float br  = b_ih[0 * HH + hc] + b_hh[0 * HH + hc];
  const float bz  = b_ih[1 * HH + hc] + b_hh[1 * HH + hc];
  const float bni = b_ih[2 * HH + hc];
  const float bnh = b_hh[2 * HH + hc];

  // --- phase C assignment: WGs 0..7 -> out tiles, 8..15 -> sel tiles ---
  const bool is_out = (wg < 8);
  const int  oct = is_out ? wg : (wg - 8);
  const int  oc  = oct * 16 + ncol;
  const float bO = is_out ? b_out[oc] : b_sel[oc];

  float local_nsel = 0.0f;

  for (int t = 0; t < TT; ++t) {
    const int p = t & 1;
    const half_t* hprev16 = hf16 + (size_t)p * BB * HH;
    const float*  hprev32 = h32  + (size_t)p * BB * HH;
    half_t* hnew16 = hf16 + (size_t)(p ^ 1) * BB * HH;
    float*  hnew32 = h32  + (size_t)(p ^ 1) * BB * HH;

    // ---------------- Phase A: gate GEMMs ----------------
    v8f ar  = splat8(br);
    v8f az  = splat8(bz);
    v8f ani = splat8(bni);
    v8f anh = splat8(bnh);

    // gi = xg @ w_ih^T  (K = F = 128), A shared across the 3 gates
#pragma unroll
    for (int kb = 0; kb < FF; kb += 32) {
      v16h a  = load_a_frag(xg16, FF, rt * 16, kb, lane);
      ar  = wmma_f16(a, load_b_frag(wih16, FF, 0 * HH + ct * 16, kb, lane), ar);
      az  = wmma_f16(a, load_b_frag(wih16, FF, 1 * HH + ct * 16, kb, lane), az);
      ani = wmma_f16(a, load_b_frag(wih16, FF, 2 * HH + ct * 16, kb, lane), ani);
    }
    // gh = h @ w_hh^T  (K = H = 256)
#pragma unroll
    for (int kb = 0; kb < HH; kb += 32) {
      v16h a  = load_a_frag(hprev16, HH, rt * 16, kb, lane);
      ar  = wmma_f16(a, load_b_frag(whh16, HH, 0 * HH + ct * 16, kb, lane), ar);
      az  = wmma_f16(a, load_b_frag(whh16, HH, 1 * HH + ct * 16, kb, lane), az);
      anh = wmma_f16(a, load_b_frag(whh16, HH, 2 * HH + ct * 16, kb, lane), anh);
    }

    // ---------------- Phase B: gates + h update ----------------
#pragma unroll
    for (int v = 0; v < 8; ++v) {
      const int row = rt * 16 + v + hi8 * 8;     // C/D layout: m = v + 8*hi8
      const int col = ct * 16 + ncol;
      const float hp = hprev32[(size_t)row * HH + col];
      const float r  = sigmoidf_(ar[v]);
      const float z  = sigmoidf_(az[v]);
      const float n  = tanhf(ani[v] + r * anh[v]);
      const float hn = (1.0f - z) * n + z * hp;
      hnew32[(size_t)row * HH + col] = hn;
      hnew16[(size_t)row * HH + col] = (half_t)hn;
    }

    grid_barrier(bcnt, bgen);   // h_new fully published

    // ---------------- Phase C: out / sel GEMMs ----------------
    v8f acc = splat8(bO);
    const half_t* Wm = is_out ? wout16 : wsel16;
#pragma unroll
    for (int kb = 0; kb < HH; kb += 32) {
      v16h a = load_a_frag(hnew16, HH, rt * 16, kb, lane);
      acc = wmma_f16(a, load_b_frag(Wm, HH, oct * 16, kb, lane), acc);
    }

    if (is_out) {
#pragma unroll
      for (int v = 0; v < 8; ++v) {
        const int row = rt * 16 + v + hi8 * 8;
        outputs[((size_t)t * BB + row) * OO + oc] = acc[v];
      }
    } else {
      // logits tile -> binary-concrete gate for step t+1 (fused, never stored)
      if (t < TT - 1) {
#pragma unroll
        for (int v = 0; v < 8; ++v) {
          const int row = rt * 16 + v + hi8 * 8;        // batch index
          const float nz = noise[((size_t)t * BB + row) * FF + oc];
          const float lg = acc[v] + __logf(nz + EPS_SEL) - __logf(1.0f - nz + EPS_SEL);
          const float w  = sigmoidf_(lg * TEMP_INV);
          selw[((size_t)(t + 1) * BB + row) * FF + oc] = w;
          local_nsel += w;
          const float xv = x[((size_t)row * TT + (t + 1)) * FF + oc];
          xg16[(size_t)row * FF + oc] = (half_t)(w * xv);
        }
      }
    }

    grid_barrier(bcnt, bgen);   // xg/selw for next step published
  }

  if (!is_out) atomicAdd(nsel, local_nsel);
}

// ---------------------------------------------------------------------------
extern "C" void kernel_launch(void* const* d_in, const int* in_sizes, int n_in,
                              void* d_out, int out_size, void* d_ws, size_t ws_size,
                              hipStream_t stream) {
  (void)in_sizes; (void)n_in; (void)out_size; (void)ws_size;
  const float* x     = (const float*)d_in[0];
  const float* noise = (const float*)d_in[1];
  const float* w_ih  = (const float*)d_in[2];
  const float* w_hh  = (const float*)d_in[3];
  const float* b_ih  = (const float*)d_in[4];
  const float* b_hh  = (const float*)d_in[5];
  const float* W_out = (const float*)d_in[6];
  const float* b_out = (const float*)d_in[7];
  const float* W_sel = (const float*)d_in[8];
  const float* b_sel = (const float*)d_in[9];
  float* out_f = (float*)d_out;

  // ---- workspace carve-out (256B aligned sections, ~1.2 MB total) ----
  char* p = (char*)d_ws;
  auto carve = [&](size_t bytes) {
    void* r = (void*)p;
    p += (bytes + 255) & ~(size_t)255;
    return r;
  };
  float*  h32    = (float*) carve((size_t)2 * BB * HH * sizeof(float));
  half_t* hf16   = (half_t*)carve((size_t)2 * BB * HH * sizeof(half_t));
  half_t* xg16   = (half_t*)carve((size_t)BB * FF * sizeof(half_t));
  half_t* wih16  = (half_t*)carve((size_t)3 * HH * FF * sizeof(half_t));
  half_t* whh16  = (half_t*)carve((size_t)3 * HH * HH * sizeof(half_t));
  half_t* wout16 = (half_t*)carve((size_t)OO * HH * sizeof(half_t));
  half_t* wsel16 = (half_t*)carve((size_t)FF * HH * sizeof(half_t));
  int*    bar    = (int*)   carve(256);

  // init covers largest region (3*H*H = 196608 elems)
  const int ithreads = 256;
  const int iblocks  = (3 * HH * HH + ithreads - 1) / ithreads;
  adagru_init<<<iblocks, ithreads, 0, stream>>>(
      x, w_ih, w_hh, W_out, W_sel,
      h32, hf16, xg16, wih16, whh16, wout16, wsel16, out_f, bar);

  adagru_persistent<<<NWG, 256, 0, stream>>>(
      x, noise, b_ih, b_hh, b_out, b_sel,
      wih16, whh16, wout16, wsel16,
      h32, hf16, xg16, out_f, bar);
}
